// mix_pcd_align_53463752901095
// MI455X (gfx1250) — matrix operand order repaired and checked by reading, not compiled, not run on hardware
//
#include <hip/hip_runtime.h>

// ---------------------------------------------------------------------------
// CDNA5 (gfx1250) PCD-alignment pipeline.
// Matrix work (conv3x3 implicit GEMM + DCN 64x576 GEMM) runs on
// v_wmma_f32_16x16x32_bf16 (wave32, 16x16 tiles), K=64 per barrier with
// double-buffered LDS so global staging overlaps the WMMAs.
// ---------------------------------------------------------------------------

typedef __attribute__((ext_vector_type(16))) __bf16 v16bf;
typedef __attribute__((ext_vector_type(8)))  __bf16 v8bf;
typedef __attribute__((ext_vector_type(8)))  float  v8f;

struct Src4 {
    const float* p[4];
    float scale[4];
};

// Pack 8 floats -> v8bf; clang lowers the pairs to v_cvt_pk_bf16_f32 and the
// store to a single 16B LDS store.
__device__ __forceinline__ v8bf pk8(const float* r) {
    v8bf h;
#pragma unroll
    for (int i = 0; i < 8; ++i) h[i] = (__bf16)r[i];
    return h;
}
__device__ __forceinline__ v8bf pk8v(float4 a, float4 b) {
    v8bf h;
    h[0] = (__bf16)a.x; h[1] = (__bf16)a.y; h[2] = (__bf16)a.z; h[3] = (__bf16)a.w;
    h[4] = (__bf16)b.x; h[5] = (__bf16)b.y; h[6] = (__bf16)b.z; h[7] = (__bf16)b.w;
    return h;
}

// Build a 16-element bf16 fragment from two 16-byte LDS chunks
// (elements 2i,2i+1 live in VGPR i per the CDNA5 WMMA register layout).
__device__ __forceinline__ v16bf load_frag(const unsigned char* lds, int off0, int off1) {
    v8bf lo = *(const v8bf*)(lds + off0);
    v8bf hi = *(const v8bf*)(lds + off1);
    v16bf r;
#pragma unroll
    for (int i = 0; i < 8; ++i) { r[i] = lo[i]; r[8 + i] = hi[i]; }
    return r;
}

// Row stride for LDS tiles: 64 bf16 (128B) padded to 144B.
// Row starts land on banks m*36 mod 64 -> 16 distinct, 4 apart: b128-conflict-free.
#define LROW 144

// ---------------------------------------------------------------------------
// conv3x3 (pad=1) as implicit GEMM.  Block: 256 thr = 8 waves (wave32).
// Block tile: 32 spatial (contiguous x in one row) x 64 output channels.
// Wave (mt,nt) owns a 16x16 D tile.  One K-step = one (source, tap) pair
// = 64 input channels = 2 WMMAs.  Double-buffered LDS, 1 barrier / step.
// Up to NSRC concatenated 64-channel sources, each with a scalar multiplier.
// ---------------------------------------------------------------------------
template <int NSRC>
__global__ __launch_bounds__(256) void conv3x3_wmma(
    Src4 src, const float* __restrict__ wgt, const float* __restrict__ bias,
    float* __restrict__ dst, int B, int H, int W, int cout, int act) {
    constexpr int NSTEPS = 9 * NSRC;
    constexpr int CIN = 64 * NSRC;
    __shared__ alignas(16) unsigned char ldsA[2][32 * LROW];
    __shared__ alignas(16) unsigned char ldsB[2][64 * LROW];

    const int tid  = threadIdx.x;
    const int lane = tid & 31;
    const int wave = tid >> 5;
    const int mt = wave >> 2;          // 0..1 spatial sub-tile
    const int nt = wave & 3;           // 0..3 channel sub-tile

    const int tilesX = W >> 5;
    int g = blockIdx.x;
    const int xt = g % tilesX; g /= tilesX;
    const int y  = g % H;
    const int b  = g / H;
    const int x0 = xt << 5;
    const int nbase = (int)blockIdx.y << 6;
    const size_t HW = (size_t)H * W;

    const float* ps[NSRC];
    float sc[NSRC];
#pragma unroll
    for (int i = 0; i < NSRC; ++i) { ps[i] = src.p[i]; sc[i] = src.scale[i]; }

    const int am  = tid & 31, ac0 = tid >> 5;   // A staging: spatial, channel group
    const int bn  = tid & 63, bk0 = tid >> 6;   // B staging: outchan, k group
    const int ngl = nbase + bn;
    const int ngc = ngl < cout ? ngl : cout - 1;
    const float nmask = ngl < cout ? 1.f : 0.f;
    const size_t bOff = (size_t)b * 64 * HW;

    v8f acc = {0.f, 0.f, 0.f, 0.f, 0.f, 0.f, 0.f, 0.f};

    auto stage = [&](int buf, int step) {
        const int s   = step / 9;
        const int tap = step - s * 9;
        const int ky = tap / 3, kx = tap - ky * 3;
        // ---- A tile: 32 spatial x 64 channels (clamped addr, masked scale).
        //      Thread owns channels [ac0*8, ac0*8+8): contiguous 16B in LDS;
        //      lanes sweep 32 consecutive x per channel -> coalesced loads.
        const int yy = y + ky - 1;
        const int xx = x0 + am + kx - 1;
        const bool ok = (yy >= 0) & (yy < H) & (xx >= 0) & (xx < W);
        const int yyc = min(max(yy, 0), H - 1);
        const int xxc = min(max(xx, 0), W - 1);
        const float scl = ok ? sc[s] : 0.f;
        const float* abase = ps[s] + bOff + (size_t)(ac0 * 8) * HW +
                             (size_t)yyc * W + xxc;
        float ra[8];
#pragma unroll
        for (int j = 0; j < 8; ++j) ra[j] = abase[(size_t)j * HW] * scl;
        *(v8bf*)(&ldsA[buf][0] + am * LROW + ac0 * 16) = pk8(ra);
        // ---- B tile: 64 outchan x 64 k (OIHW gather, stride 9 over k).
        //      Thread owns k in [bk0*16, bk0*16+16): contiguous 32B in LDS.
        const float* wbase = wgt + ((size_t)ngc * CIN + s * 64 + bk0 * 16) * 9 +
                             ky * 3 + kx;
        float rw[16];
#pragma unroll
        for (int j = 0; j < 16; ++j) rw[j] = wbase[(size_t)j * 9] * nmask;
        unsigned char* Bdst = &ldsB[buf][0] + bn * LROW + bk0 * 32;
        *(v8bf*)(Bdst)      = pk8(rw);
        *(v8bf*)(Bdst + 16) = pk8(rw + 8);
    };

    stage(0, 0);
    for (int step = 0; step < NSTEPS; ++step) {
        __syncthreads();
        if (step + 1 < NSTEPS) stage((step + 1) & 1, step + 1);
        if (step + 2 < NSTEPS) {       // warm WGP$/L2 for the tile after next
            const int s2   = (step + 2) / 9;
            const int tap2 = (step + 2) - s2 * 9;
            const int ky2 = tap2 / 3, kx2 = tap2 - ky2 * 3;
            const int yy2 = min(max(y + ky2 - 1, 0), H - 1);
            const int xx2 = min(max(x0 + am + kx2 - 1, 0), W - 1);
            __builtin_prefetch(ps[s2] + bOff + (size_t)(ac0 * 8) * HW +
                               (size_t)yy2 * W + xx2, 0, 3);
        }
        const unsigned char* A  = &ldsA[step & 1][0];
        const unsigned char* Bq = &ldsB[step & 1][0];
        const int mrow = (mt << 4) + (lane & 15);
        const int nrow = (nt << 4) + (lane & 15);
        const int hi   = lane >> 4;
#pragma unroll
        for (int c = 0; c < 2; ++c) {   // two K=32 WMMAs per 64-deep step
            v16bf af = load_frag(A, mrow * LROW + c * 64 + hi * 16,
                                 mrow * LROW + c * 64 + 32 + hi * 16);
            v16bf bf = load_frag(Bq, nrow * LROW + c * 64 + hi * 16,
                                 nrow * LROW + c * 64 + 32 + hi * 16);
            acc = __builtin_amdgcn_wmma_f32_16x16x32_bf16(
                false, af, false, bf, (short)0, acc, false, false);
        }
    }

    // ---- epilogue: bias + optional leaky ReLU, scatter to NCHW ----
    const int ncol = nbase + (nt << 4) + (lane & 15);
    if (ncol < cout) {
        const float bv = bias[ncol];
        const int mbase = (mt << 4) + ((lane >> 4) << 3);
#pragma unroll
        for (int r = 0; r < 8; ++r) {
            float v = acc[r] + bv;
            if (act) v = (v >= 0.f) ? v : 0.1f * v;
            dst[((size_t)b * cout + ncol) * HW + (size_t)y * W + (x0 + mbase + r)] = v;
        }
    }
}

// ---------------------------------------------------------------------------
// DCN second stage: D[p][64] = val[p][576] x W[64][576]^T + bias.
// val is pre-im2col'd bf16 (k = cin*9 + tap, identical to flat conv.w order).
// 9 K-steps of 64; double-buffered; A copied with b128s, B via float4 + pk8.
// ---------------------------------------------------------------------------
__global__ __launch_bounds__(256) void dcn_gemm_wmma(
    const unsigned short* __restrict__ valA, const float* __restrict__ wgt,
    const float* __restrict__ bias, float* __restrict__ dst,
    int B, int H, int W, int act) {
    __shared__ alignas(16) unsigned char ldsA[2][32 * LROW];
    __shared__ alignas(16) unsigned char ldsB[2][64 * LROW];

    const int tid  = threadIdx.x;
    const int lane = tid & 31;
    const int wave = tid >> 5;
    const int mt = wave >> 2, nt = wave & 3;
    const int p0 = (int)blockIdx.x << 5;

    v8f acc = {0.f, 0.f, 0.f, 0.f, 0.f, 0.f, 0.f, 0.f};

    const int am = tid >> 3, akq = tid & 7;     // A: 32 rows x 16B chunks
    const int bn = tid & 63, bkq = tid >> 6;    // B: 64 rows x 16-float slice
    const unsigned short* arow = valA + (size_t)(p0 + am) * 576 + akq * 8;
    const float* wrow = wgt + (size_t)bn * 576 + bkq * 16;

    auto stage = [&](int buf, int step) {
        *(int4*)(&ldsA[buf][0] + am * LROW + akq * 16) =
            *(const int4*)(arow + step * 64);
        const float* wb = wrow + step * 64;
        float4 w0 = *(const float4*)(wb + 0);
        float4 w1 = *(const float4*)(wb + 4);
        float4 w2 = *(const float4*)(wb + 8);
        float4 w3 = *(const float4*)(wb + 12);
        unsigned char* Bdst = &ldsB[buf][0] + bn * LROW + bkq * 32;
        *(v8bf*)(Bdst)      = pk8v(w0, w1);
        *(v8bf*)(Bdst + 16) = pk8v(w2, w3);
    };

    stage(0, 0);
    for (int step = 0; step < 9; ++step) {      // 576 = 9 * 64
        __syncthreads();
        if (step + 1 < 9) stage((step + 1) & 1, step + 1);
        if (step + 2 < 9) __builtin_prefetch(arow + (step + 2) * 64, 0, 3);
        const unsigned char* A  = &ldsA[step & 1][0];
        const unsigned char* Bq = &ldsB[step & 1][0];
        const int mrow = (mt << 4) + (lane & 15);
        const int nrow = (nt << 4) + (lane & 15);
        const int hi   = lane >> 4;
#pragma unroll
        for (int c = 0; c < 2; ++c) {
            v16bf af = load_frag(A, mrow * LROW + c * 64 + hi * 16,
                                 mrow * LROW + c * 64 + 32 + hi * 16);
            v16bf bf = load_frag(Bq, nrow * LROW + c * 64 + hi * 16,
                                 nrow * LROW + c * 64 + 32 + hi * 16);
            acc = __builtin_amdgcn_wmma_f32_16x16x32_bf16(
                false, af, false, bf, (short)0, acc, false, false);
        }
    }

    const size_t HW = (size_t)H * W;
    const int xb = p0 % W;
    const int yb = (p0 / W) % H;
    const int bb = p0 / (W * H);
    const int ncol  = (nt << 4) + (lane & 15);
    const int mbase = (mt << 4) + ((lane >> 4) << 3);
    const float bv = bias[ncol];
#pragma unroll
    for (int r = 0; r < 8; ++r) {
        float v = acc[r] + bv;
        if (act) v = (v >= 0.f) ? v : 0.1f * v;
        dst[((size_t)bb * 64 + ncol) * HW + (size_t)yb * W + (xb + mbase + r)] = v;
    }
}

// ---------------------------------------------------------------------------
// DCNv2 sampling: offsets/mask -> masked bilinear im2col val[p][576] (bf16).
// One thread per (b, group, tap, y, x); 8 channels per group.
// ---------------------------------------------------------------------------
__global__ void dcn_sample_kernel(const float* __restrict__ x, const float* __restrict__ om,
                                  unsigned short* __restrict__ val, int B, int H, int W) {
    const int total = B * 72 * H * W;
    int idx = (int)blockIdx.x * 256 + (int)threadIdx.x;
    if (idx >= total) return;
    int t = idx;
    const int xx = t % W; t /= W;
    const int yy = t % H; t /= H;
    const int kk = t % 9; t /= 9;
    const int gg = t % 8;
    const int b  = t / 8;

    const size_t HW  = (size_t)H * W;
    const size_t pix = (size_t)yy * W + xx;
    const float* omb = om + (size_t)b * 216 * HW + pix;
    const int och = gg * 9 + kk;
    const float dy = omb[(size_t)och * HW];
    const float dx = omb[(size_t)(72 + och) * HW];
    const float ml = omb[(size_t)(144 + och) * HW];
    const float mask = 1.f / (1.f + __expf(-ml));

    const float py = dy + (float)(kk / 3 - 1) + (float)yy;
    const float px = dx + (float)(kk % 3 - 1) + (float)xx;
    const float y0f = floorf(py), x0f = floorf(px);
    const int iy0 = (int)y0f, ix0 = (int)x0f;
    const float wy = py - y0f, wx = px - x0f;
    const bool vy0 = (iy0 >= 0) & (iy0 < H);
    const bool vy1 = (iy0 + 1 >= 0) & (iy0 + 1 < H);
    const bool vx0 = (ix0 >= 0) & (ix0 < W);
    const bool vx1 = (ix0 + 1 >= 0) & (ix0 + 1 < W);
    const float w00 = (vy0 && vx0) ? (1.f - wy) * (1.f - wx) : 0.f;
    const float w01 = (vy0 && vx1) ? (1.f - wy) * wx : 0.f;
    const float w10 = (vy1 && vx0) ? wy * (1.f - wx) : 0.f;
    const float w11 = (vy1 && vx1) ? wy * wx : 0.f;
    const int cy0 = min(max(iy0, 0), H - 1), cy1 = min(max(iy0 + 1, 0), H - 1);
    const int cx0 = min(max(ix0, 0), W - 1), cx1 = min(max(ix0 + 1, 0), W - 1);

    const float* xb = x + ((size_t)b * 64 + gg * 8) * HW;
    unsigned short* op = val + (((size_t)b * H + yy) * W + xx) * 576 + (size_t)gg * 72 + kk;
#pragma unroll
    for (int c = 0; c < 8; ++c) {
        const float* xc = xb + (size_t)c * HW;
        float s = w00 * xc[(size_t)cy0 * W + cx0] + w01 * xc[(size_t)cy0 * W + cx1]
                + w10 * xc[(size_t)cy1 * W + cx0] + w11 * xc[(size_t)cy1 * W + cx1];
        *(__bf16*)(op + c * 9) = (__bf16)(s * mask);
    }
}

// ---------------------------------------------------------------------------
// Bilinear 2x upsample (align_corners=False, half-pixel centers, edge clamp).
// ---------------------------------------------------------------------------
__global__ void up2_bilinear(const float* __restrict__ in, float* __restrict__ out,
                             int B, int C, int h, int w) {
    const int W2 = 2 * w, H2 = 2 * h;
    const int total = B * C * H2 * W2;
    int idx = (int)blockIdx.x * 256 + (int)threadIdx.x;
    if (idx >= total) return;
    int t = idx;
    const int ox = t % W2; t /= W2;
    const int oy = t % H2; t /= H2;
    const int c  = t % C;
    const int b  = t / C;

    const float sy = (oy + 0.5f) * 0.5f - 0.5f;
    const float sx = (ox + 0.5f) * 0.5f - 0.5f;
    const float fy0 = floorf(sy), fx0 = floorf(sx);
    const float wy = sy - fy0, wx = sx - fx0;
    const int y0 = (int)fy0, x0 = (int)fx0;
    const int y0c = min(max(y0, 0), h - 1), y1c = min(max(y0 + 1, 0), h - 1);
    const int x0c = min(max(x0, 0), w - 1), x1c = min(max(x0 + 1, 0), w - 1);

    const float* base = in + ((size_t)b * C + c) * h * w;
    const float v = (1.f - wy) * ((1.f - wx) * base[(size_t)y0c * w + x0c] +
                                  wx * base[(size_t)y0c * w + x1c]) +
                    wy * ((1.f - wx) * base[(size_t)y1c * w + x0c] +
                          wx * base[(size_t)y1c * w + x1c]);
    out[idx] = v;
}

// ---------------------------------------------------------------------------
// Host orchestration.
// d_in order = jax tree-flatten of setup_inputs(): sorted dict keys at every
// level. Top: nbr_l1, nbr_l2, nbr_l3, params..., ref_l1, ref_l2, ref_l3.
// conv entries flatten b,w; dcn entries flatten conv.b, conv.w, om.b, om.w.
// ---------------------------------------------------------------------------
enum {
    IN_NBR_L1 = 0, IN_NBR_L2, IN_NBR_L3,
    P_L1_DCN_CB, P_L1_DCN_CW, P_L1_DCN_OMB, P_L1_DCN_OMW,                 // 3..6
    P_L1_DCN_L2_CB, P_L1_DCN_L2_CW, P_L1_DCN_L2_OMB, P_L1_DCN_L2_OMW,     // 7..10
    P_L1_DCN_L3_CB, P_L1_DCN_L3_CW, P_L1_DCN_L3_OMB, P_L1_DCN_L3_OMW,     // 11..14
    P_L1_OC1_B, P_L1_OC1_W,
    P_L1_OC1_L2_B, P_L1_OC1_L2_W,
    P_L1_OC1_L3_B, P_L1_OC1_L3_W,
    P_L1_OC2_B, P_L1_OC2_W,
    P_L1_OC2_L2_B, P_L1_OC2_L2_W,
    P_L1_OC2_L3_B, P_L1_OC2_L3_W,
    P_L1_OC3_B, P_L1_OC3_W,
    P_L1_OC3_L2_B, P_L1_OC3_L2_W,
    P_L1_OC3_L3_B, P_L1_OC3_L3_W,                                          // ..32
    P_L2_DCN_CB, P_L2_DCN_CW, P_L2_DCN_OMB, P_L2_DCN_OMW,                  // 33..36 (unused, faithful)
    P_L2_DCN_L3_CB, P_L2_DCN_L3_CW, P_L2_DCN_L3_OMB, P_L2_DCN_L3_OMW,      // 37..40
    P_L2_OC1_B, P_L2_OC1_W,
    P_L2_OC1_L3_B, P_L2_OC1_L3_W,
    P_L2_OC2_B, P_L2_OC2_W,
    P_L2_OC2_L3_B, P_L2_OC2_L3_W,
    P_L2_OC3_B, P_L2_OC3_W,
    P_L2_OC3_L3_B, P_L2_OC3_L3_W,                                          // ..52
    P_L3_DCN_CB, P_L3_DCN_CW, P_L3_DCN_OMB, P_L3_DCN_OMW,                  // 53..56
    P_L3_OC1_B, P_L3_OC1_W,
    P_L3_OC2_B, P_L3_OC2_W,                                                // ..60
    P_CAS_DCN_CB, P_CAS_DCN_CW, P_CAS_DCN_OMB, P_CAS_DCN_OMW,              // 61..64
    P_CAS_OC1_B, P_CAS_OC1_W,
    P_CAS_OC2_B, P_CAS_OC2_W,                                              // ..68
    P_FUSE_B, P_FUSE_W,                                                    // 69,70
    IN_REF_L1, IN_REF_L2, IN_REF_L3                                        // 71..73
};

extern "C" void kernel_launch(void* const* d_in, const int* in_sizes, int n_in,
                              void* d_out, int out_size, void* d_ws, size_t ws_size,
                              hipStream_t stream) {
    (void)in_sizes; (void)n_in; (void)out_size; (void)ws_size;

    const int B = 4, NF = 64;
    const int H1 = 128, W1 = 128, H2 = 64, W2 = 64, H3 = 32, W3 = 32;

    auto in = [&](int i) { return (const float*)d_in[i]; };

    // ---- workspace bump allocator ----
    char* wsb = (char*)d_ws;
    size_t off = 0;
    auto alloc = [&](size_t bytes) -> char* {
        char* p = wsb + off;
        off = (off + bytes + 255) & ~(size_t)255;
        return p;
    };
    const size_t F1B = (size_t)B * NF * H1 * W1 * 4;
    const size_t F2B = (size_t)B * NF * H2 * W2 * 4;
    const size_t F3B = (size_t)B * NF * H3 * W3 * 4;
    const size_t OM1B = (size_t)B * 216 * H1 * W1 * 4;
    const size_t OM2B = (size_t)B * 216 * H2 * W2 * 4;
    const size_t OM3B = (size_t)B * 216 * H3 * W3 * 4;
    const size_t V1B = (size_t)B * H1 * W1 * 576 * 2;
    const size_t V2B = (size_t)B * H2 * W2 * 576 * 2;
    const size_t V3B = (size_t)B * H3 * W3 * 576 * 2;

    float* t3a     = (float*)alloc(F3B);
    float* L3off   = (float*)alloc(F3B);
    float* L3fea   = (float*)alloc(F3B);
    float* om3     = (float*)alloc(OM3B);
    unsigned short* val3 = (unsigned short*)alloc(V3B);

    float* L3offup = (float*)alloc(F2B);
    float* nbr3up  = (float*)alloc(F2B);
    float* t2      = (float*)alloc(F2B);
    float* u2      = (float*)alloc(F2B);
    float* L2offl3 = (float*)alloc(F2B);
    float* L2feal3 = (float*)alloc(F2B);
    float* L2off   = (float*)alloc(F2B);
    float* L2fea   = (float*)alloc(F2B);
    float* om2     = (float*)alloc(OM2B);
    unsigned short* val2 = (unsigned short*)alloc(V2B);

    float* L2offup   = (float*)alloc(F1B);
    float* L1feal3up = (float*)alloc(F1B);
    float* t1        = (float*)alloc(F1B);
    float* u1        = (float*)alloc(F1B);
    float* L1offl3   = (float*)alloc(F1B);
    float* L1feal3   = (float*)alloc(F1B);
    float* L1feal2up = (float*)alloc(F1B);
    float* L1offl2   = (float*)alloc(F1B);
    float* L1feal2   = (float*)alloc(F1B);
    float* L1off     = (float*)alloc(F1B);
    float* L1fea     = (float*)alloc(F1B);
    float* fuseb     = (float*)alloc(F1B);
    float* om1       = (float*)alloc(OM1B);
    unsigned short* val1 = (unsigned short*)alloc(V1B);

    float* outFea = (float*)d_out;
    float* outOff = outFea + (size_t)B * NF * H1 * W1;

    // ---- launch helpers ----
    auto conv = [&](Src4 s, int nsrc, int iw, int ib, float* dst,
                    int H, int W, int cout, int act) {
        dim3 grid((unsigned)(B * H * (W / 32)), (unsigned)((cout + 63) / 64));
        dim3 blk(256);
        switch (nsrc) {
        case 1: conv3x3_wmma<1><<<grid, blk, 0, stream>>>(s, in(iw), in(ib), dst, B, H, W, cout, act); break;
        case 2: conv3x3_wmma<2><<<grid, blk, 0, stream>>>(s, in(iw), in(ib), dst, B, H, W, cout, act); break;
        case 3: conv3x3_wmma<3><<<grid, blk, 0, stream>>>(s, in(iw), in(ib), dst, B, H, W, cout, act); break;
        default: conv3x3_wmma<4><<<grid, blk, 0, stream>>>(s, in(iw), in(ib), dst, B, H, W, cout, act); break;
        }
    };
    auto S1 = [](const float* a) {
        Src4 s{}; s.p[0] = a; s.scale[0] = 1.f; return s;
    };
    auto S2 = [](const float* a, const float* b, float sb) {
        Src4 s{}; s.p[0] = a; s.scale[0] = 1.f; s.p[1] = b; s.scale[1] = sb; return s;
    };
    auto S3 = [](const float* a, const float* b, const float* c, float sc) {
        Src4 s{}; s.p[0] = a; s.scale[0] = 1.f; s.p[1] = b; s.scale[1] = 1.f;
        s.p[2] = c; s.scale[2] = sc; return s;
    };
    auto S4 = [](const float* a, const float* b, const float* c, const float* d, float sd) {
        Src4 s{}; s.p[0] = a; s.scale[0] = 1.f; s.p[1] = b; s.scale[1] = 1.f;
        s.p[2] = c; s.scale[2] = 1.f; s.p[3] = d; s.scale[3] = sd; return s;
    };
    auto sample = [&](const float* x, const float* om, unsigned short* val, int H, int W) {
        int total = B * 72 * H * W;
        dcn_sample_kernel<<<dim3((unsigned)((total + 255) / 256)), dim3(256), 0, stream>>>(
            x, om, val, B, H, W);
    };
    auto gemm = [&](const unsigned short* val, int iw, int ib, float* dst,
                    int H, int W, int act) {
        dcn_gemm_wmma<<<dim3((unsigned)((B * H * W) / 32)), dim3(256), 0, stream>>>(
            val, in(iw), in(ib), dst, B, H, W, act);
    };
    auto up2 = [&](const float* src, float* dst, int h, int w) {
        int total = B * NF * 4 * h * w;
        up2_bilinear<<<dim3((unsigned)((total + 255) / 256)), dim3(256), 0, stream>>>(
            src, dst, B, NF, h, w);
    };

    const float* nbr1 = in(IN_NBR_L1); const float* nbr2 = in(IN_NBR_L2);
    const float* nbr3 = in(IN_NBR_L3);
    const float* ref1 = in(IN_REF_L1); const float* ref2 = in(IN_REF_L2);
    const float* ref3 = in(IN_REF_L3);

    // ==================== L3 ====================
    conv(S2(nbr3, ref3, 1.f), 2, P_L3_OC1_W, P_L3_OC1_B, t3a, H3, W3, 64, 1);
    conv(S1(t3a), 1, P_L3_OC2_W, P_L3_OC2_B, L3off, H3, W3, 64, 1);
    conv(S1(L3off), 1, P_L3_DCN_OMW, P_L3_DCN_OMB, om3, H3, W3, 216, 0);
    sample(nbr3, om3, val3, H3, W3);
    gemm(val3, P_L3_DCN_CW, P_L3_DCN_CB, L3fea, H3, W3, 1);      // lrelu(dcn)
    up2(L3off, L3offup, H3, W3);
    up2(L3fea, nbr3up, H3, W3);

    // ==================== L2 ====================
    conv(S2(nbr3up, ref2, 1.f), 2, P_L2_OC1_L3_W, P_L2_OC1_L3_B, t2, H2, W2, 64, 1);
    conv(S2(t2, L3offup, 2.f), 2, P_L2_OC2_L3_W, P_L2_OC2_L3_B, u2, H2, W2, 64, 1);
    conv(S1(u2), 1, P_L2_OC3_L3_W, P_L2_OC3_L3_B, L2offl3, H2, W2, 64, 1);
    conv(S1(L2offl3), 1, P_L2_DCN_L3_OMW, P_L2_DCN_L3_OMB, om2, H2, W2, 216, 0);
    sample(nbr3up, om2, val2, H2, W2);
    gemm(val2, P_L2_DCN_L3_CW, P_L2_DCN_L3_CB, L2feal3, H2, W2, 0);
    conv(S3(nbr2, L2feal3, ref2, 1.f), 3, P_L2_OC1_W, P_L2_OC1_B, t2, H2, W2, 64, 1);
    conv(S3(t2, L2offl3, L3offup, 2.f), 3, P_L2_OC2_W, P_L2_OC2_B, u2, H2, W2, 64, 1);
    conv(S1(u2), 1, P_L2_OC3_W, P_L2_OC3_B, L2off, H2, W2, 64, 1);
    // faithful to reference: L2_dcn_l3 params reused for the second L2 DCN
    conv(S1(L2off), 1, P_L2_DCN_L3_OMW, P_L2_DCN_L3_OMB, om2, H2, W2, 216, 0);
    sample(nbr2, om2, val2, H2, W2);
    gemm(val2, P_L2_DCN_L3_CW, P_L2_DCN_L3_CB, L2fea, H2, W2, 0);

    // ==================== L1 ====================
    up2(L2off, L2offup, H2, W2);
    up2(L2feal3, L1feal3up, H2, W2);
    conv(S2(L1feal3up, ref1, 1.f), 2, P_L1_OC1_L3_W, P_L1_OC1_L3_B, t1, H1, W1, 64, 1);
    conv(S2(t1, L2offup, 2.f), 2, P_L1_OC2_L3_W, P_L1_OC2_L3_B, u1, H1, W1, 64, 1);
    conv(S1(u1), 1, P_L1_OC3_L3_W, P_L1_OC3_L3_B, L1offl3, H1, W1, 64, 1);
    conv(S1(L1offl3), 1, P_L1_DCN_L3_OMW, P_L1_DCN_L3_OMB, om1, H1, W1, 216, 0);
    sample(L1feal3up, om1, val1, H1, W1);
    gemm(val1, P_L1_DCN_L3_CW, P_L1_DCN_L3_CB, L1feal3, H1, W1, 0);

    up2(L2fea, L1feal2up, H2, W2);
    conv(S3(L1feal2up, L1feal3, ref1, 1.f), 3, P_L1_OC1_L2_W, P_L1_OC1_L2_B, t1, H1, W1, 64, 1);
    conv(S3(t1, L1offl3, L2offup, 2.f), 3, P_L1_OC2_L2_W, P_L1_OC2_L2_B, u1, H1, W1, 64, 1);
    conv(S1(u1), 1, P_L1_OC3_L2_W, P_L1_OC3_L2_B, L1offl2, H1, W1, 64, 1);
    conv(S1(L1offl2), 1, P_L1_DCN_L2_OMW, P_L1_DCN_L2_OMB, om1, H1, W1, 216, 0);
    sample(L1feal2up, om1, val1, H1, W1);
    gemm(val1, P_L1_DCN_L2_CW, P_L1_DCN_L2_CB, L1feal2, H1, W1, 0);

    conv(S4(nbr1, L1feal3, L1feal2, ref1, 1.f), 4, P_L1_OC1_W, P_L1_OC1_B, t1, H1, W1, 64, 1);
    conv(S4(t1, L1offl2, L1offl3, L2offup, 2.f), 4, P_L1_OC2_W, P_L1_OC2_B, u1, H1, W1, 64, 1);
    conv(S1(u1), 1, P_L1_OC3_W, P_L1_OC3_B, L1off, H1, W1, 64, 1);
    conv(S1(L1off), 1, P_L1_DCN_OMW, P_L1_DCN_OMB, om1, H1, W1, 216, 0);
    sample(nbr1, om1, val1, H1, W1);
    gemm(val1, P_L1_DCN_CW, P_L1_DCN_CB, L1fea, H1, W1, 0);

    // ==================== cascade ====================
    conv(S4(L1fea, L1feal3, L1feal2, ref1, 1.f), 4, P_CAS_OC1_W, P_CAS_OC1_B, t1, H1, W1, 64, 1);
    conv(S1(t1), 1, P_CAS_OC2_W, P_CAS_OC2_B, outOff, H1, W1, 64, 1);   // -> offset output
    conv(S3(L1fea, L1feal3, L1feal2, 1.f), 3, P_FUSE_W, P_FUSE_B, fuseb, H1, W1, 64, 0);
    conv(S1(outOff), 1, P_CAS_DCN_OMW, P_CAS_DCN_OMB, om1, H1, W1, 216, 0);
    sample(fuseb, om1, val1, H1, W1);
    gemm(val1, P_CAS_DCN_CW, P_CAS_DCN_CB, outFea, H1, W1, 1);          // lrelu -> fea output
}
